// RowSelfAttention_28836410426043
// MI455X (gfx1250) — matrix operand order repaired and checked by reading, not compiled, not run on hardware
//
#include <hip/hip_runtime.h>
#include <hip/hip_bf16.h>

typedef __attribute__((ext_vector_type(16))) __bf16 v16bf;
typedef __attribute__((ext_vector_type(8)))  __bf16 v8bf;
typedef __attribute__((ext_vector_type(8)))  float  v8f;
typedef __attribute__((ext_vector_type(4)))  float  v4f;

#define R_DIM 64
#define I_DIM 512
#define N_DIM 4
#define E_DIM 256
#define H_DIM 8
#define D_DIM 32
#define RD_DIM (R_DIM * D_DIM)             // 2048
#define M_DIM  (R_DIM * I_DIM * N_DIM)     // 131072
#define HN_DIM (H_DIM * N_DIM)             // 32
// scaling = D^-0.5 / sqrt(R) = (1/sqrt(32))/8
#define Q_SCALE 0.02209708691207961f

// ---------- fragment helpers (layouts per CDNA5 ISA §7.12.2) ----------

static __device__ inline v16bf cat16(v8bf lo, v8bf hi) {
  v16bf r;
#pragma unroll
  for (int t = 0; t < 8; ++t) { r[t] = lo[t]; r[t + 8] = hi[t]; }
  return r;
}

// A fragment (16x32 bf16, row-major source, ld = row stride in elements).
// lane-half 0: K chunks [k0..k0+7],[k0+16..k0+23]; half 1: [k0+8..15],[k0+24..31]
static __device__ inline v16bf ldA_bf(const __bf16* base, long ld, int row, int k0, int half) {
  const __bf16* p = base + (long)row * ld + k0 + half * 8;
  return cat16(*(const v8bf*)p, *(const v8bf*)(p + 16));
}

// B fragment (32x16 bf16). Source stored as B^T row-major: BT[col][k], ld = K stride.
// lane-half 0: K chunks [k0..k0+7],[k0+16..k0+23]; half 1: [k0+8..15],[k0+24..31]
static __device__ inline v16bf ldB_bf(const __bf16* base, long ld, int col, int k0, int half) {
  const __bf16* p = base + (long)col * ld + k0 + half * 8;
  return cat16(*(const v8bf*)p, *(const v8bf*)(p + 16));
}

static __device__ inline v8f wmma_bf16(v16bf a, v16bf b, v8f c) {
  return __builtin_amdgcn_wmma_f32_16x16x32_bf16(false, a, false, b, (short)0, c, false, false);
}

// ---------- kernel 0a: convert weights fp32 -> bf16 (W row-major == B^T layout) ----------
__global__ void k_prep_weights(const float* __restrict__ Wq, const float* __restrict__ Wk,
                               const float* __restrict__ Wv, const float* __restrict__ Wo,
                               __bf16* __restrict__ Wbf) {
  int gid = blockIdx.x * blockDim.x + threadIdx.x;
  if (gid >= 4 * E_DIM * E_DIM) return;
  int which = gid >> 16;          // 65536 elements per matrix
  int off   = gid & 65535;
  const float* src = (which == 0) ? Wq : (which == 1) ? Wk : (which == 2) ? Wv : Wo;
  Wbf[gid] = (__bf16)src[off];
}

// ---------- kernel 0b: convert x fp32 -> bf16, 8 elements/thread ----------
__global__ void k_prep_x(const float* __restrict__ x, __bf16* __restrict__ xbf) {
  long gid = (long)blockIdx.x * blockDim.x + threadIdx.x;  // M*E/8 threads
  const float* p = x + gid * 8;
  v4f a = *(const v4f*)p;
  v4f b = *(const v4f*)(p + 4);
  v8bf r;
  r[0] = (__bf16)a[0]; r[1] = (__bf16)a[1]; r[2] = (__bf16)a[2]; r[3] = (__bf16)a[3];
  r[4] = (__bf16)b[0]; r[5] = (__bf16)b[1]; r[6] = (__bf16)b[2]; r[7] = (__bf16)b[3];
  *(v8bf*)(xbf + gid * 8) = r;
}

// ---------- kernel 1: fused Q/K/V projections, 32x64 wave tile, all-bf16 path ----------
// q/k -> Qp/Kp [hn][i][r*D+d] bf16 ; v -> VpT [hn][r*D+d][i] bf16
__global__ void k_proj(const __bf16* __restrict__ xbf, const __bf16* __restrict__ Wbf,
                       const float* __restrict__ bq, const float* __restrict__ bk,
                       const float* __restrict__ bv,
                       __bf16* __restrict__ Qp, __bf16* __restrict__ Kp,
                       __bf16* __restrict__ VpT) {
  const int lane = threadIdx.x & 31;
  const int wv   = threadIdx.x >> 5;
  const int which = blockIdx.y;               // 0=q 1=k 2=v
  const int tile  = blockIdx.x * 8 + wv;      // 16384 wave tiles (4096 x 4)
  const int TM = M_DIM / 32;                  // 4096
  const int tile_m = tile % TM;
  const int tile_f = tile / TM;               // 0..3
  const int m0 = tile_m * 32, f0 = tile_f * 64;
  const int L = lane & 15, half = lane >> 4;

  const __bf16* Wb   = Wbf + (long)which * E_DIM * E_DIM;
  const float*  bias = (which == 0) ? bq : (which == 1) ? bk : bv;

  v8f acc[2][4] = {};
#pragma unroll 2
  for (int k0 = 0; k0 < E_DIM; k0 += 32) {
    v16bf a[2], b[4];
#pragma unroll
    for (int im = 0; im < 2; ++im)
      a[im] = ldA_bf(xbf, E_DIM, m0 + im * 16 + L, k0, half);
#pragma unroll
    for (int jn = 0; jn < 4; ++jn)
      b[jn] = ldB_bf(Wb, E_DIM, f0 + jn * 16 + L, k0, half);
#pragma unroll
    for (int im = 0; im < 2; ++im)
#pragma unroll
      for (int jn = 0; jn < 4; ++jn)
        acc[im][jn] = wmma_bf16(a[im], b[jn], acc[im][jn]);
  }

  const float scale = (which == 0) ? Q_SCALE : 1.0f;
#pragma unroll
  for (int jn = 0; jn < 4; ++jn) {
    const int f = f0 + jn * 16 + L;
    const int h = f >> 5, d = f & 31;
    const float bval = bias[f];
#pragma unroll
    for (int im = 0; im < 2; ++im) {
#pragma unroll
      for (int g = 0; g < 8; ++g) {
        int m = m0 + im * 16 + g + half * 8;
        int n = m & (N_DIM - 1);
        int i = (m >> 2) & (I_DIM - 1);
        int r = m >> 11;                      // m / (I*N)
        float val = (acc[im][jn][g] + bval) * scale;
        long hn = (long)h * N_DIM + n;
        if (which == 2) {
          VpT[(hn * RD_DIM + r * D_DIM + d) * (long)I_DIM + i] = (__bf16)val;
        } else {
          __bf16* dst = (which == 0) ? Qp : Kp;
          dst[(hn * I_DIM + i) * (long)RD_DIM + r * D_DIM + d] = (__bf16)val;
        }
      }
    }
  }
}

// ---------- kernel 2: logits S = Qp * Kp^T + T5 bucket bias, 32x64 wave tile ----------
__global__ void k_logits(const __bf16* __restrict__ Qp, const __bf16* __restrict__ Kp,
                         const int* __restrict__ distances, const float* __restrict__ rel_emb,
                         float* __restrict__ logits) {
  const int lane = threadIdx.x & 31;
  const int wv   = threadIdx.x >> 5;
  const int hn   = blockIdx.y;                 // 0..31
  const int tile = blockIdx.x * 8 + wv;        // 128 tiles per hn (16 x 8)
  const int tile_i = tile & 15, tile_j = tile >> 4;
  const int i0 = tile_i * 32, j0 = tile_j * 64;
  const int L = lane & 15, half = lane >> 4;

  const __bf16* Qb = Qp + (long)hn * I_DIM * RD_DIM;
  const __bf16* Kb = Kp + (long)hn * I_DIM * RD_DIM;

  v8f acc[2][4] = {};
#pragma unroll 2
  for (int k0 = 0; k0 < RD_DIM; k0 += 32) {    // 64 steps x 8 WMMAs
    v16bf a[2], b[4];
#pragma unroll
    for (int im = 0; im < 2; ++im)
      a[im] = ldA_bf(Qb, RD_DIM, i0 + im * 16 + L, k0, half);
#pragma unroll
    for (int jn = 0; jn < 4; ++jn)
      b[jn] = ldB_bf(Kb, RD_DIM, j0 + jn * 16 + L, k0, half);
#pragma unroll
    for (int im = 0; im < 2; ++im)
#pragma unroll
      for (int jn = 0; jn < 4; ++jn)
        acc[im][jn] = wmma_bf16(a[im], b[jn], acc[im][jn]);
  }

  const int h = hn >> 2, n = hn & 3;
#pragma unroll
  for (int jn = 0; jn < 4; ++jn) {
    const int j = j0 + jn * 16 + L;
#pragma unroll
    for (int im = 0; im < 2; ++im) {
#pragma unroll
      for (int g = 0; g < 8; ++g) {
        int i = i0 + im * 16 + g + half * 8;
        int dist = distances[((long)n * I_DIM + i) * I_DIM + j];
        float ad = fabsf((float)dist);
        // searchsorted-left on linspace(0, 50000, 63): bucket = #boundaries < ad
        int bucket = (ad <= 0.0f) ? 0 : (int)ceilf(ad * (62.0f / 50000.0f));
        bucket = (bucket > 63) ? 63 : bucket;
        float bias = rel_emb[bucket * H_DIM + h];
        logits[((long)hn * I_DIM + i) * I_DIM + j] = acc[im][jn][g] + bias;
      }
    }
  }
}

// ---------- kernel 3: in-place softmax over j; also emit bf16 copy ----------
__global__ void k_softmax(float* __restrict__ probs, __bf16* __restrict__ Pbf) {
  const int lane = threadIdx.x & 31;
  const int wv   = threadIdx.x >> 5;
  const long row = (long)blockIdx.x * 8 + wv;  // 16384 rows of length 512
  float* p = probs + row * I_DIM;

  float vals[16];
  float mx = -3.402823466e38f;
#pragma unroll
  for (int t = 0; t < 16; ++t) { vals[t] = p[lane + t * 32]; mx = fmaxf(mx, vals[t]); }
#pragma unroll
  for (int s = 16; s >= 1; s >>= 1) mx = fmaxf(mx, __shfl_xor(mx, s, 32));

  float sum = 0.0f;
#pragma unroll
  for (int t = 0; t < 16; ++t) { vals[t] = __expf(vals[t] - mx); sum += vals[t]; }
#pragma unroll
  for (int s = 16; s >= 1; s >>= 1) sum += __shfl_xor(sum, s, 32);
  float inv = 1.0f / sum;

  __bf16* pb = Pbf + row * I_DIM;
#pragma unroll
  for (int t = 0; t < 16; ++t) {
    float v = vals[t] * inv;
    p[lane + t * 32]  = v;
    pb[lane + t * 32] = (__bf16)v;
  }
}

// ---------- kernel 4: context C[i, r*D+d] = P * VpT, 32x64 wave tile ----------
__global__ void k_context(const __bf16* __restrict__ Pbf, const __bf16* __restrict__ VpT,
                          __bf16* __restrict__ ctx) {
  const int lane = threadIdx.x & 31;
  const int wv   = threadIdx.x >> 5;
  const int hn   = blockIdx.y;
  const int tile = blockIdx.x * 8 + wv;        // 512 tiles per hn (16 x 32)
  const int tile_i = tile & 15, tile_rd = tile >> 4;
  const int i0 = tile_i * 32, rd0 = tile_rd * 64;
  const int L = lane & 15, half = lane >> 4;

  const __bf16* Pb = Pbf + (long)hn * I_DIM * I_DIM;
  const __bf16* Vb = VpT + (long)hn * RD_DIM * I_DIM;

  v8f acc[2][4] = {};
#pragma unroll 2
  for (int k0 = 0; k0 < I_DIM; k0 += 32) {     // 16 steps x 8 WMMAs
    v16bf a[2], b[4];
#pragma unroll
    for (int im = 0; im < 2; ++im)
      a[im] = ldA_bf(Pb, I_DIM, i0 + im * 16 + L, k0, half);
#pragma unroll
    for (int jn = 0; jn < 4; ++jn)
      b[jn] = ldB_bf(Vb, I_DIM, rd0 + jn * 16 + L, k0, half);
#pragma unroll
    for (int im = 0; im < 2; ++im)
#pragma unroll
      for (int jn = 0; jn < 4; ++jn)
        acc[im][jn] = wmma_bf16(a[im], b[jn], acc[im][jn]);
  }

  const int h = hn >> 2, n = hn & 3;
#pragma unroll
  for (int jn = 0; jn < 4; ++jn) {
    const int rd = rd0 + jn * 16 + L;
    const int r = rd >> 5, d = rd & 31;
    const int e = h * D_DIM + d;
#pragma unroll
    for (int im = 0; im < 2; ++im) {
#pragma unroll
      for (int g = 0; g < 8; ++g) {
        int i = i0 + im * 16 + g + half * 8;
        long m = ((long)r * I_DIM + i) * N_DIM + n;
        ctx[m * E_DIM + e] = (__bf16)acc[im][jn][g];
      }
    }
  }
}

// ---------- kernel 5: output projection out[m,f] = ctx * Wo^T + bo, 32x64 wave tile ----------
__global__ void k_outproj(const __bf16* __restrict__ ctx, const __bf16* __restrict__ WoBF,
                          const float* __restrict__ bo, float* __restrict__ out) {
  const int lane = threadIdx.x & 31;
  const int wv   = threadIdx.x >> 5;
  const int tile = blockIdx.x * 8 + wv;        // 16384 wave tiles (4096 x 4)
  const int TM = M_DIM / 32;                   // 4096
  const int tile_m = tile % TM, tile_f = tile / TM;
  const int m0 = tile_m * 32, f0 = tile_f * 64;
  const int L = lane & 15, half = lane >> 4;

  v8f acc[2][4] = {};
#pragma unroll 2
  for (int k0 = 0; k0 < E_DIM; k0 += 32) {
    v16bf a[2], b[4];
#pragma unroll
    for (int im = 0; im < 2; ++im)
      a[im] = ldA_bf(ctx, E_DIM, m0 + im * 16 + L, k0, half);
#pragma unroll
    for (int jn = 0; jn < 4; ++jn)
      b[jn] = ldB_bf(WoBF, E_DIM, f0 + jn * 16 + L, k0, half);
#pragma unroll
    for (int im = 0; im < 2; ++im)
#pragma unroll
      for (int jn = 0; jn < 4; ++jn)
        acc[im][jn] = wmma_bf16(a[im], b[jn], acc[im][jn]);
  }

#pragma unroll
  for (int jn = 0; jn < 4; ++jn) {
    const int f = f0 + jn * 16 + L;
    const float bval = bo[f];
#pragma unroll
    for (int im = 0; im < 2; ++im) {
#pragma unroll
      for (int g = 0; g < 8; ++g) {
        long m = m0 + im * 16 + g + half * 8;
        out[m * E_DIM + f] = acc[im][jn][g] + bval;
      }
    }
  }
}

// ---------- launch ----------
extern "C" void kernel_launch(void* const* d_in, const int* in_sizes, int n_in,
                              void* d_out, int out_size, void* d_ws, size_t ws_size,
                              hipStream_t stream) {
  (void)in_sizes; (void)n_in; (void)out_size; (void)ws_size;

  const float* x         = (const float*)d_in[0];
  const int*   distances = (const int*)d_in[1];
  const float* Wq        = (const float*)d_in[2];
  const float* bq        = (const float*)d_in[3];
  const float* Wk        = (const float*)d_in[4];
  const float* bk        = (const float*)d_in[5];
  const float* Wv        = (const float*)d_in[6];
  const float* bv        = (const float*)d_in[7];
  const float* Wo        = (const float*)d_in[8];
  const float* bo        = (const float*)d_in[9];
  const float* rel_emb   = (const float*)d_in[10];

  float* out   = (float*)d_out;
  float* probs = out + (long)M_DIM * E_DIM;    // attn_probs region of d_out (HN*I*I fp32)

  // workspace layout (bytes)
  char* ws = (char*)d_ws;
  const long SZ_W   = 4L * E_DIM * E_DIM * 2;            //  512 KB
  const long SZ_X   = (long)M_DIM * E_DIM * 2;           //   64 MB
  const long SZ_QKV = (long)HN_DIM * I_DIM * RD_DIM * 2; //   64 MB each
  const long SZ_P   = (long)HN_DIM * I_DIM * I_DIM * 2;  //   16 MB
  __bf16* Wbf = (__bf16*)(ws);
  __bf16* xbf = (__bf16*)(ws + SZ_W);
  __bf16* Qp  = (__bf16*)(ws + SZ_W + SZ_X);
  __bf16* Kp  = (__bf16*)(ws + SZ_W + SZ_X + SZ_QKV);
  __bf16* VpT = (__bf16*)(ws + SZ_W + SZ_X + 2 * SZ_QKV);
  __bf16* Pbf = (__bf16*)(ws + SZ_W + SZ_X + 3 * SZ_QKV);
  __bf16* ctx = (__bf16*)(ws + SZ_W + SZ_X + 3 * SZ_QKV + SZ_P);

  hipLaunchKernelGGL(k_prep_weights, dim3(1024), dim3(256), 0, stream, Wq, Wk, Wv, Wo, Wbf);
  hipLaunchKernelGGL(k_prep_x, dim3((M_DIM * E_DIM / 8) / 256), dim3(256), 0, stream, x, xbf);
  hipLaunchKernelGGL(k_proj,    dim3(2048, 3), dim3(256), 0, stream,
                     xbf, Wbf, bq, bk, bv, Qp, Kp, VpT);
  hipLaunchKernelGGL(k_logits,  dim3(16, 32),  dim3(256), 0, stream,
                     Qp, Kp, distances, rel_emb, probs);
  hipLaunchKernelGGL(k_softmax, dim3(2048),    dim3(256), 0, stream, probs, Pbf);
  hipLaunchKernelGGL(k_context, dim3(64, 32),  dim3(256), 0, stream, Pbf, VpT, ctx);
  hipLaunchKernelGGL(k_outproj, dim3(2048),    dim3(256), 0, stream,
                     ctx, Wbf + 3L * E_DIM * E_DIM, bo, out);
}
